// Memory_Block_58420145160845
// MI455X (gfx1250) — matrix operand order repaired and checked by reading, not compiled, not run on hardware
//
#include <hip/hip_runtime.h>

// ---------------------------------------------------------------------------
// MI455X / gfx1250 implementation of the VQ soft-codebook block.
//   score GEMMs + output GEMM run on v_wmma_f32_16x16x32_f16 (f32 accum).
//   GEMM1 fuses row-argmax (embed_ind); GEMM2 fuses score2 store + exp +
//   row-sum-of-exp (softmax denominator); GEMM3 applies 1/rowsum in epilogue.
// ---------------------------------------------------------------------------

typedef __attribute__((ext_vector_type(16))) _Float16 v16h;
typedef __attribute__((ext_vector_type(8)))  _Float16 f16x8;
typedef __attribute__((ext_vector_type(8)))  float    v8f;

union FragU { v16h v; f16x8 h[2]; };

#define C_   512     // channels (contraction dim of score GEMMs)
#define HW_  4096    // H*W
#define N_   32768   // B*H*W rows
#define K_   1024    // NUMS codebook entries (contraction dim of out GEMM)
#define VC_  512     // V_C output channels

// order-preserving float->uint map for packed (score,col) max
__device__ __forceinline__ unsigned long long pack_max(float f, unsigned c) {
  unsigned u = __float_as_uint(f);
  u = (u & 0x80000000u) ? ~u : (u | 0x80000000u);
  return ((unsigned long long)u << 32) | (unsigned long long)c;
}

// A fragment (16x32 f16): lane holds row = base+lane%16; elems 0..15 -> K=16*half+0..15
// contiguous => two b128 loads at +0 and +16 bytes.
__device__ __forceinline__ void load_fragA(FragU& f, const _Float16* p) {
  f.h[0] = *(const f16x8*)p;
  f.h[1] = *(const f16x8*)(p + 8);
}
// B fragment (32x16 f16): lane holds col = base+lane%16; elems 0..15 -> K=16*half+0..15
// contiguous => two b128 loads at +0 and +16 bytes.
__device__ __forceinline__ void load_fragB(FragU& f, const _Float16* p) {
  f.h[0] = *(const f16x8*)p;
  f.h[1] = *(const f16x8*)(p + 8);
}

// ---------------------------------------------------------------------------
__global__ void zero_f32(float* __restrict__ p, int n) {
  int i = blockIdx.x * 256 + threadIdx.x;
  if (i < n) p[i] = 0.f;
}

// (B,C,H,W) -> (N,C) transpose + row L2-normalize -> f16. 16 rows/block via LDS.
__global__ __launch_bounds__(256) void xn_norm_kernel(const float* __restrict__ x,
                                                      _Float16* __restrict__ xn) {
  __shared__ float tile[16 * 513];
  __shared__ float rsc[16];
  const int tid = threadIdx.x;
  const int n0  = blockIdx.x * 16;
  const int b   = n0 >> 12;
  const int hw0 = n0 & 4095;
  const float* xb = x + (size_t)b * C_ * HW_;
  #pragma unroll 4
  for (int it = 0; it < 32; ++it) {              // 16*512 elems / 256 threads
    int idx = it * 256 + tid;
    int c = idx >> 4, hl = idx & 15;             // lanes 0..15 coalesce 64B
    tile[hl * 513 + c] = xb[(size_t)c * HW_ + hw0 + hl];
  }
  __syncthreads();
  {
    int row = tid >> 4, sub = tid & 15;          // 16 threads per row
    float s = 0.f;
    for (int c = sub; c < C_; c += 16) { float v = tile[row * 513 + c]; s += v * v; }
    s += __shfl_xor(s, 1, 32); s += __shfl_xor(s, 2, 32);
    s += __shfl_xor(s, 4, 32); s += __shfl_xor(s, 8, 32);
    if (sub == 0) rsc[row] = 1.0f / fmaxf(sqrtf(s), 1e-12f);
  }
  __syncthreads();
  #pragma unroll 4
  for (int it = 0; it < 32; ++it) {
    int idx = it * 256 + tid;
    int row = idx >> 9, c = idx & 511;           // coalesced f16 row-major store
    xn[(size_t)(n0 + row) * C_ + c] = (_Float16)(tile[row * 513 + c] * rsc[row]);
  }
}

// one block per row: L2-normalize a (rows x 512) f32 matrix row into f16
__global__ __launch_bounds__(256) void rownorm_f16_kernel(const float* __restrict__ src,
                                                          _Float16* __restrict__ dst) {
  __shared__ float red[8];
  __shared__ float ssc;
  const int r = blockIdx.x, tid = threadIdx.x;
  const float* p = src + (size_t)r * C_;
  float s = 0.f;
  for (int c = tid; c < C_; c += 256) { float v = p[c]; s += v * v; }
  for (int m = 16; m >= 1; m >>= 1) s += __shfl_xor(s, m, 32);
  if ((tid & 31) == 0) red[tid >> 5] = s;
  __syncthreads();
  if (tid == 0) {
    float t = 0.f;
    for (int i = 0; i < 8; ++i) t += red[i];
    ssc = 1.0f / fmaxf(sqrtf(t), 1e-12f);
  }
  __syncthreads();
  float sc = ssc;
  for (int c = tid; c < C_; c += 256) dst[(size_t)r * C_ + c] = (_Float16)(p[c] * sc);
}

// GEMM1: score = xn(32768x512) * mn(1024x512)^T, fused per-row argmax -> embed_ind
__global__ __launch_bounds__(256) void gemm_score_argmax(const _Float16* __restrict__ Xn,
                                                         const _Float16* __restrict__ Mn,
                                                         int* __restrict__ embed_ind) {
  __shared__ unsigned long long part[4][64];
  __shared__ unsigned long long best[64];
  const int tid = threadIdx.x;
  if (tid < 64) best[tid] = 0ull;
  const int lane = tid & 31, w = tid >> 5;
  const int half = lane >> 4, l16 = lane & 15;
  const int waveRow = w >> 2, waveCol = w & 3;
  const int rowBase = blockIdx.x * 64 + waveRow * 32;

  for (int cc = 0; cc < K_; cc += 256) {
    const int colBase = cc + waveCol * 64;
    v8f acc[2][4];
    #pragma unroll
    for (int tr = 0; tr < 2; ++tr)
      #pragma unroll
      for (int tc = 0; tc < 4; ++tc)
        #pragma unroll
        for (int e = 0; e < 8; ++e) acc[tr][tc][e] = 0.f;

    for (int k0 = 0; k0 < C_; k0 += 32) {
      FragU a[2], bf[4];
      #pragma unroll
      for (int tr = 0; tr < 2; ++tr)
        load_fragA(a[tr], Xn + (size_t)(rowBase + tr * 16 + l16) * C_ + k0 + 16 * half);
      #pragma unroll
      for (int tc = 0; tc < 4; ++tc)
        load_fragB(bf[tc], Mn + (size_t)(colBase + tc * 16 + l16) * C_ + k0 + 16 * half);
      #pragma unroll
      for (int tr = 0; tr < 2; ++tr)
        #pragma unroll
        for (int tc = 0; tc < 4; ++tc)
          acc[tr][tc] = __builtin_amdgcn_wmma_f32_16x16x32_f16(
              false, a[tr].v, false, bf[tc].v, (short)0, acc[tr][tc], false, false);
    }

    // per-row argmax over this 256-col chunk (C/D layout: col=lane%16, row=v+8*half)
    #pragma unroll
    for (int tr = 0; tr < 2; ++tr) {
      #pragma unroll
      for (int e = 0; e < 8; ++e) {
        float bv = acc[tr][0][e];
        unsigned bc = (unsigned)(colBase + l16);
        #pragma unroll
        for (int tc = 1; tc < 4; ++tc) {
          float v = acc[tr][tc][e];
          unsigned c = (unsigned)(colBase + tc * 16 + l16);
          if (v > bv) { bv = v; bc = c; }
        }
        unsigned long long pk = pack_max(bv, bc);
        #pragma unroll
        for (int m = 1; m < 16; m <<= 1) {
          unsigned long long o = __shfl_xor(pk, m, 32);
          if (o > pk) pk = o;
        }
        if (l16 == 0) part[waveCol][waveRow * 32 + tr * 16 + 8 * half + e] = pk;
      }
    }
    __syncthreads();
    if (tid < 64) {
      unsigned long long b0 = best[tid];
      #pragma unroll
      for (int j = 0; j < 4; ++j) if (part[j][tid] > b0) b0 = part[j][tid];
      best[tid] = b0;
    }
    __syncthreads();
  }
  if (tid < 64) embed_ind[blockIdx.x * 64 + tid] = (int)(best[tid] & 0xffffffffull);
}

// segment-sum scatter: esum[ind[n],:] += xf[n,:] ; counts[ind[n]] += 1
__global__ __launch_bounds__(256) void scatter_kernel(const float* __restrict__ x,
                                                      const int* __restrict__ ind,
                                                      float* __restrict__ esum,
                                                      float* __restrict__ counts) {
  const int n = blockIdx.x * 256 + threadIdx.x;
  const int b = n >> 12, hw = n & 4095;
  const int k = ind[n];
  const float* xb = x + (size_t)b * C_ * HW_ + hw;
  float* er = esum + (size_t)k * C_;
  for (int c = 0; c < C_; ++c) atomicAdd(&er[c], xb[(size_t)c * HW_]);
  atomicAdd(&counts[k], 1.0f);
}

// new_m = m*rate + (esum/(cnt+1e-6))*(1-rate); mn2 = normalize(new_m) -> f16
__global__ __launch_bounds__(256) void newm_norm_kernel(const float* __restrict__ m,
                                                        const float* __restrict__ esum,
                                                        const float* __restrict__ counts,
                                                        _Float16* __restrict__ mn2) {
  __shared__ float row[C_];
  __shared__ float red[8];
  __shared__ float ssc;
  const int k = blockIdx.x, tid = threadIdx.x;
  const float invc = 1.0f / (counts[k] + 1e-6f);
  const float rate = 0.999f, omr = 1.0f - 0.999f;
  float s = 0.f;
  for (int c = tid; c < C_; c += 256) {
    float v = m[(size_t)k * C_ + c] * rate + esum[(size_t)k * C_ + c] * invc * omr;
    row[c] = v; s += v * v;
  }
  for (int mm = 16; mm >= 1; mm >>= 1) s += __shfl_xor(s, mm, 32);
  if ((tid & 31) == 0) red[tid >> 5] = s;
  __syncthreads();
  if (tid == 0) {
    float t = 0.f;
    for (int i = 0; i < 8; ++i) t += red[i];
    ssc = 1.0f / fmaxf(sqrtf(t), 1e-12f);
  }
  __syncthreads();
  float sc = ssc;
  for (int c = tid; c < C_; c += 256) mn2[(size_t)k * C_ + c] = (_Float16)(row[c] * sc);
}

// GEMM2: score2 = xn * mn2^T; write score2 f32, expP=exp(score2) f16, rowsum=sum(exp)
__global__ __launch_bounds__(256) void gemm_score2_softprep(const _Float16* __restrict__ Xn,
                                                            const _Float16* __restrict__ Mn2,
                                                            float* __restrict__ score2,
                                                            _Float16* __restrict__ expP,
                                                            float* __restrict__ rowsum) {
  __shared__ float part[4][64];
  __shared__ float rs[64];
  const int tid = threadIdx.x;
  if (tid < 64) rs[tid] = 0.f;
  const int lane = tid & 31, w = tid >> 5;
  const int half = lane >> 4, l16 = lane & 15;
  const int waveRow = w >> 2, waveCol = w & 3;
  const int rowBase = blockIdx.x * 64 + waveRow * 32;

  for (int cc = 0; cc < K_; cc += 256) {
    const int colBase = cc + waveCol * 64;
    v8f acc[2][4];
    #pragma unroll
    for (int tr = 0; tr < 2; ++tr)
      #pragma unroll
      for (int tc = 0; tc < 4; ++tc)
        #pragma unroll
        for (int e = 0; e < 8; ++e) acc[tr][tc][e] = 0.f;

    for (int k0 = 0; k0 < C_; k0 += 32) {
      FragU a[2], bf[4];
      #pragma unroll
      for (int tr = 0; tr < 2; ++tr)
        load_fragA(a[tr], Xn + (size_t)(rowBase + tr * 16 + l16) * C_ + k0 + 16 * half);
      #pragma unroll
      for (int tc = 0; tc < 4; ++tc)
        load_fragB(bf[tc], Mn2 + (size_t)(colBase + tc * 16 + l16) * C_ + k0 + 16 * half);
      #pragma unroll
      for (int tr = 0; tr < 2; ++tr)
        #pragma unroll
        for (int tc = 0; tc < 4; ++tc)
          acc[tr][tc] = __builtin_amdgcn_wmma_f32_16x16x32_f16(
              false, a[tr].v, false, bf[tc].v, (short)0, acc[tr][tc], false, false);
    }

    #pragma unroll
    for (int tr = 0; tr < 2; ++tr) {
      #pragma unroll
      for (int e = 0; e < 8; ++e) {
        const int n = rowBase + tr * 16 + 8 * half + e;
        float se = 0.f;
        #pragma unroll
        for (int tc = 0; tc < 4; ++tc) {
          float v = acc[tr][tc][e];
          int c = colBase + tc * 16 + l16;
          score2[(size_t)n * K_ + c] = v;
          float ev = __expf(v);                  // |score|<=~1, no max-shift needed
          expP[(size_t)n * K_ + c] = (_Float16)ev;
          se += ev;
        }
        se += __shfl_xor(se, 1, 32); se += __shfl_xor(se, 2, 32);
        se += __shfl_xor(se, 4, 32); se += __shfl_xor(se, 8, 32);
        if (l16 == 0) part[waveCol][waveRow * 32 + tr * 16 + 8 * half + e] = se;
      }
    }
    __syncthreads();
    if (tid < 64) rs[tid] += part[0][tid] + part[1][tid] + part[2][tid] + part[3][tid];
    __syncthreads();
  }
  if (tid < 64) rowsum[blockIdx.x * 64 + tid] = rs[tid];
}

// std (1024x512) -> stdT (512x1024) f16 via LDS tile transpose
__global__ __launch_bounds__(256) void transpose_std_kernel(const float* __restrict__ stdv,
                                                            _Float16* __restrict__ stdT) {
  __shared__ float t[32][33];
  const int tx = threadIdx.x & 31, ty = threadIdx.x >> 5;  // 0..7
  const int k0 = blockIdx.x * 32, v0 = blockIdx.y * 32;
  #pragma unroll
  for (int i = 0; i < 32; i += 8) t[ty + i][tx] = stdv[(size_t)(k0 + ty + i) * VC_ + v0 + tx];
  __syncthreads();
  #pragma unroll
  for (int i = 0; i < 32; i += 8)
    stdT[(size_t)(v0 + ty + i) * K_ + k0 + tx] = (_Float16)t[tx][ty + i];
}

// GEMM3: out = (expP * stdT^T) * (1/rowsum[n])   (softmax @ std)
__global__ __launch_bounds__(256) void gemm_out_kernel(const _Float16* __restrict__ P,
                                                       const _Float16* __restrict__ StdT,
                                                       const float* __restrict__ rowsum,
                                                       float* __restrict__ out) {
  const int tid = threadIdx.x;
  const int lane = tid & 31, w = tid >> 5;
  const int half = lane >> 4, l16 = lane & 15;
  const int waveRow = w >> 2, waveCol = w & 3;
  const int rowBase = blockIdx.x * 64 + waveRow * 32;

  for (int cc = 0; cc < VC_; cc += 256) {
    const int colBase = cc + waveCol * 64;
    v8f acc[2][4];
    #pragma unroll
    for (int tr = 0; tr < 2; ++tr)
      #pragma unroll
      for (int tc = 0; tc < 4; ++tc)
        #pragma unroll
        for (int e = 0; e < 8; ++e) acc[tr][tc][e] = 0.f;

    for (int k0 = 0; k0 < K_; k0 += 32) {
      FragU a[2], bf[4];
      #pragma unroll
      for (int tr = 0; tr < 2; ++tr)
        load_fragA(a[tr], P + (size_t)(rowBase + tr * 16 + l16) * K_ + k0 + 16 * half);
      #pragma unroll
      for (int tc = 0; tc < 4; ++tc)
        load_fragB(bf[tc], StdT + (size_t)(colBase + tc * 16 + l16) * K_ + k0 + 16 * half);
      #pragma unroll
      for (int tr = 0; tr < 2; ++tr)
        #pragma unroll
        for (int tc = 0; tc < 4; ++tc)
          acc[tr][tc] = __builtin_amdgcn_wmma_f32_16x16x32_f16(
              false, a[tr].v, false, bf[tc].v, (short)0, acc[tr][tc], false, false);
    }

    #pragma unroll
    for (int tr = 0; tr < 2; ++tr) {
      #pragma unroll
      for (int e = 0; e < 8; ++e) {
        const int n = rowBase + tr * 16 + 8 * half + e;
        const float inv = 1.0f / rowsum[n];
        #pragma unroll
        for (int tc = 0; tc < 4; ++tc)
          out[(size_t)n * VC_ + colBase + tc * 16 + l16] = acc[tr][tc][e] * inv;
      }
    }
  }
}

// ---------------------------------------------------------------------------
extern "C" void kernel_launch(void* const* d_in, const int* in_sizes, int n_in,
                              void* d_out, int out_size, void* d_ws, size_t ws_size,
                              hipStream_t stream) {
  (void)in_sizes; (void)n_in; (void)out_size; (void)ws_size;
  const float* x    = (const float*)d_in[0];   // (8,512,64,64) f32
  const float* m    = (const float*)d_in[1];   // (1024,512)   f32
  const float* stdv = (const float*)d_in[2];   // (1024,512)   f32

  float* out    = (float*)d_out;               // (N, VC) f32
  float* score2 = out + (size_t)N_ * VC_;      // (N, K)  f32

  char* wsp = (char*)d_ws;
  _Float16* xn   = (_Float16*)wsp;  wsp += (size_t)N_ * C_ * 2;   // 32 MB
  _Float16* mn   = (_Float16*)wsp;  wsp += (size_t)K_ * C_ * 2;   // 1 MB
  _Float16* mn2  = (_Float16*)wsp;  wsp += (size_t)K_ * C_ * 2;   // 1 MB
  _Float16* stdT = (_Float16*)wsp;  wsp += (size_t)VC_ * K_ * 2;  // 1 MB
  _Float16* expP = (_Float16*)wsp;  wsp += (size_t)N_ * K_ * 2;   // 64 MB
  float* esum    = (float*)wsp;     wsp += (size_t)K_ * C_ * 4;   // 2 MB
  float* counts  = (float*)wsp;     wsp += (size_t)K_ * 4;        // contiguous after esum
  float* rowsum  = (float*)wsp;     wsp += (size_t)N_ * 4;
  int* embed_ind = (int*)wsp;       wsp += (size_t)N_ * 4;

  zero_f32<<<(K_ * C_ + K_ + 255) / 256, 256, 0, stream>>>(esum, K_ * C_ + K_);
  xn_norm_kernel<<<N_ / 16, 256, 0, stream>>>(x, xn);
  rownorm_f16_kernel<<<K_, 256, 0, stream>>>(m, mn);
  gemm_score_argmax<<<N_ / 64, 256, 0, stream>>>(xn, mn, embed_ind);
  scatter_kernel<<<N_ / 256, 256, 0, stream>>>(x, embed_ind, esum, counts);
  newm_norm_kernel<<<K_, 256, 0, stream>>>(m, esum, counts, mn2);
  gemm_score2_softprep<<<N_ / 64, 256, 0, stream>>>(xn, mn2, score2, expP, rowsum);
  transpose_std_kernel<<<dim3(K_ / 32, VC_ / 32), 256, 0, stream>>>(stdv, stdT);
  gemm_out_kernel<<<N_ / 64, 256, 0, stream>>>(expP, stdT, rowsum, out);
}
